// V5VectorQuantizer_59897613910104
// MI455X (gfx1250) — matrix operand order repaired and checked by reading, not compiled, not run on hardware
//
#include <hip/hip_runtime.h>
#include <hip/hip_bf16.h>

typedef __attribute__((ext_vector_type(16))) _Float16 v16h;
typedef __attribute__((ext_vector_type(8)))  _Float16 h8v;
typedef __attribute__((ext_vector_type(4)))  _Float16 h4v;
typedef __attribute__((ext_vector_type(8)))  float    v8f;
typedef int v4i __attribute__((vector_size(4 * sizeof(int))));

#define NK 8
#define NS 1024
#define ND 64
#define NM 8192
#define LDIM 512

// output offsets (floats): symbol_ids, hard_q, st_q, assignment_probs, 4 scalars
#define O_SYM    0ull
#define O_HARD   65536ull
#define O_ST     4259840ull
#define O_PROBS  8454144ull
#define O_SCAL   75563008ull

// workspace offsets (floats)
#define W_AVG  0
#define W_SQ   8192
#define W_CSQ  8704
#define W_H16  16896   // f16 row-major codebook copy   (524288 halves = 262144 floats)
#define W_T16  279040  // f16 chunk-transposed codebook (524288 halves)

// -2 / ln(2): scores kept in log2 domain so v_exp_f32 (native exp2) needs no scaling
#define NEG2_INVLN2 (-2.8853900817779268f)

#if defined(__has_builtin)
#if __has_builtin(__builtin_amdgcn_global_load_async_to_lds_b128) && \
    __has_builtin(__builtin_amdgcn_s_wait_asynccnt)
#define HAVE_ASYNC 1
#endif
#endif
#ifndef HAVE_ASYNC
#define HAVE_ASYNC 0
#endif

#define TO_LDS(p) ((__attribute__((address_space(3))) v4i*)(unsigned int)(uintptr_t)(p))
#define TO_GLB(p) ((__attribute__((address_space(1))) v4i*)(uintptr_t)(p))

__device__ __forceinline__ void atomAddGlobal(float* p, float v) {
  __hip_atomic_fetch_add(p, v, __ATOMIC_RELAXED, __HIP_MEMORY_SCOPE_AGENT);
}
__device__ __forceinline__ void atomAddShared(float* p, float v) {
  __hip_atomic_fetch_add(p, v, __ATOMIC_RELAXED, __HIP_MEMORY_SCOPE_WORKGROUP);
}

__global__ void vq_zero(float* __restrict__ ws) {
  int i = blockIdx.x * 256 + threadIdx.x;
  if (i <= W_SQ) ws[i] = 0.f;
}

// prepass: |c|^2 per codeword + f16 copies (row-major and chunk-transposed)
__global__ __launch_bounds__(256) void vq_csq(const float* __restrict__ cbk,
                                              float* __restrict__ csq,
                                              _Float16* __restrict__ cbh,
                                              _Float16* __restrict__ cbt) {
  int rI = blockIdx.x * 256 + threadIdx.x;
  if (rI >= NK * NS) return;
  const int k = rI >> 10, s = rI & 1023, c = s >> 5, sl = s & 31;
  const float4* p = (const float4*)(cbk + (size_t)rI * ND);
  _Float16* hrow = cbh + (size_t)rI * ND;
  _Float16* tblk = cbt + (size_t)(k * 32 + c) * 2048 + sl;   // [d][32] within chunk
  float ssum = 0.f;
#pragma unroll
  for (int i = 0; i < 16; ++i) {
    float4 v = p[i];
    ssum += v.x * v.x + v.y * v.y + v.z * v.z + v.w * v.w;
    h4v h = {(_Float16)v.x, (_Float16)v.y, (_Float16)v.z, (_Float16)v.w};
    *(h4v*)&hrow[i * 4] = h;
    tblk[(i * 4 + 0) * 32] = h[0]; tblk[(i * 4 + 1) * 32] = h[1];
    tblk[(i * 4 + 2) * 32] = h[2]; tblk[(i * 4 + 3) * 32] = h[3];
  }
  csq[rI] = ssum;
}

// load a v16h B-fragment: 16 consecutive f16 (two 16B LDS reads)
__device__ __forceinline__ v16h load_h16(const _Float16* p) {
  union { v16h v; h8v h[2]; } u;
  u.h[0] = *(const h8v*)p;
  u.h[1] = *(const h8v*)(p + 8);
  return u.v;
}

#if HAVE_ASYNC
// wave 0, all 32 lanes: async-copy one 4KB f16 chunk into padded LDS rows.
template <bool WITH_CT>
__device__ __forceinline__ void issue_chunk(const _Float16* __restrict__ cbh_k,
                                            const _Float16* __restrict__ cbt_k,
                                            _Float16 (*Ch)[72], _Float16 (*Ct)[40],
                                            int c, int lane) {
  const _Float16* sh = cbh_k + (size_t)c * 2048;
#pragma unroll
  for (int t = 0; t < 8; ++t) {
    int idx = t * 32 + lane;
    __builtin_amdgcn_global_load_async_to_lds_b128(
        TO_GLB(sh + idx * 8), TO_LDS(&Ch[idx >> 3][(idx & 7) * 8]), 0, 0);
  }
  if (WITH_CT) {
    const _Float16* st = cbt_k + (size_t)c * 2048;
#pragma unroll
    for (int t = 0; t < 8; ++t) {
      int idx = t * 32 + lane;
      __builtin_amdgcn_global_load_async_to_lds_b128(
          TO_GLB(st + idx * 8), TO_LDS(&Ct[idx >> 2][(idx & 3) * 8]), 0, 0);
    }
  }
}
#endif

// Workgroup: 256 threads = 8 waves; wave w handles rows m0..m0+15 of codebook k.
__global__ __launch_bounds__(256) void vq_main(const float* __restrict__ lat,
                                               const float* __restrict__ mask,
                                               const float* __restrict__ cbk,
                                               const float* __restrict__ csq,
                                               const _Float16* __restrict__ cbh,
                                               const _Float16* __restrict__ cbt,
                                               float* __restrict__ out,
                                               float* __restrict__ ws) {
  __shared__ __align__(16) _Float16 C_h[2][32][72];  // chunk f16, row-major (144B rows)
  __shared__ __align__(16) _Float16 C_t[2][64][40];  // chunk f16, transposed (80B rows)
  __shared__ __align__(16) _Float16 p_st[8][16][40]; // per-wave f16 P stage
  __shared__ float avg_lds[1024];
  __shared__ float lds_sq;

  const int tid  = threadIdx.x;
  const int wave = tid >> 5, lane = tid & 31;
  const int r    = lane & 15, half = lane >> 4;
  const int k    = blockIdx.x >> 6;
  const int mt   = blockIdx.x & 63;
  const int m0   = mt * 128 + wave * 16;

  for (int i = tid; i < 1024; i += 256) avg_lds[i] = 0.f;
  if (tid == 0) lds_sq = 0.f;

  // ---- latent A-fragments (f16) + per-row |x|^2 ----
  const float* xrow = lat + (size_t)(m0 + r) * LDIM + k * ND;
  v16h afrag[2];
  float xs = 0.f;
#pragma unroll
  for (int kt = 0; kt < 2; ++kt)
#pragma unroll
    for (int sg = 0; sg < 2; ++sg) {
      const float4* p4 = (const float4*)(xrow + kt * 32 + 16 * half + sg * 8);
      float4 x0 = p4[0], x1 = p4[1];
      float e[8] = {x0.x, x0.y, x0.z, x0.w, x1.x, x1.y, x1.z, x1.w};
#pragma unroll
      for (int j = 0; j < 8; ++j) {
        afrag[kt][sg * 8 + j] = (_Float16)e[j];
        xs += e[j] * e[j];
      }
    }
  xs += __shfl_xor(xs, 16, 32);
  float maskr = mask[m0 + r];
  float xsq_v[8], mask_v[8];
#pragma unroll
  for (int v = 0; v < 8; ++v) {
    xsq_v[v]  = __shfl(xs, v + 8 * half, 32);
    mask_v[v] = __shfl(maskr, v + 8 * half, 32);
  }

  float lmax2[8], lsum[8], lmind2[8];
  int   lminid[8];
#pragma unroll
  for (int v = 0; v < 8; ++v) { lmax2[v] = -1.0e38f; lsum[v] = 0.f; lmind2[v] = 3.0e38f; lminid[v] = 0; }

  const float* cbbase = cbk + (size_t)k * NS * ND;
  const float* csbase = csq + k * NS;
  const _Float16* cbh_k = cbh + (size_t)k * 65536;
  const _Float16* cbt_k = cbt + (size_t)k * 65536;
  (void)cbbase; (void)cbh_k; (void)cbt_k;

  // ================= sweep 1: distances -> lane-local max2 / sumexp2 / argmin ========
#if HAVE_ASYNC
  if (wave == 0) issue_chunk<false>(cbh_k, cbt_k, C_h[0], C_t[0], 0, lane);
#endif
  for (int c = 0; c < 32; ++c) {
#if HAVE_ASYNC
    __syncthreads();                     // everyone done with buf[(c+1)&1]
    if (wave == 0) {
      if (c < 31) {
        issue_chunk<false>(cbh_k, cbt_k, C_h[(c + 1) & 1], C_t[(c + 1) & 1], c + 1, lane);
        __builtin_amdgcn_s_wait_asynccnt(8);   // in-order: chunk c has landed
      } else {
        __builtin_amdgcn_s_wait_asynccnt(0);
      }
    }
    __syncthreads();                     // publish chunk c
    _Float16 (*Ch)[72] = C_h[c & 1];
#else
    __syncthreads();
    for (int t = tid; t < 512; t += 256) {
      int row = t >> 4, c4 = (t & 15) * 4;
      float4 v = *(const float4*)(cbbase + (size_t)(c * 32 + row) * ND + c4);
      h4v h = {(_Float16)v.x, (_Float16)v.y, (_Float16)v.z, (_Float16)v.w};
      *(h4v*)&C_h[0][row][c4] = h;
    }
    __syncthreads();
    _Float16 (*Ch)[72] = C_h[0];
#endif
#pragma unroll
    for (int nt = 0; nt < 2; ++nt) {
      v8f acc = {};
#pragma unroll
      for (int kt = 0; kt < 2; ++kt) {
        v16h b = load_h16(&Ch[nt * 16 + r][kt * 32 + 16 * half]);
        acc = __builtin_amdgcn_wmma_f32_16x16x32_f16(false, afrag[kt], false, b,
                                                     (short)0, acc, false, false);
      }
      const int s0 = c * 32 + nt * 16;
      const float cs = csbase[s0 + r];
#pragma unroll
      for (int v = 0; v < 8; ++v) {
        float d2 = xsq_v[v] + cs - 2.0f * acc[v];
        bool bt = d2 < lmind2[v];
        lmind2[v] = bt ? d2 : lmind2[v];
        lminid[v] = bt ? (s0 + r) : lminid[v];
        float sc = NEG2_INVLN2 * d2;
        float nm = fmaxf(lmax2[v], sc);
        lsum[v] = lsum[v] * __builtin_amdgcn_exp2f(lmax2[v] - nm)
                + __builtin_amdgcn_exp2f(sc - nm);
        lmax2[v] = nm;
      }
    }
  }

  // ---- single cross-lane merge (within each 16-lane group) ----
  float rowmax2[8], inv_rs[8];
  int   minid[8];
#pragma unroll
  for (int v = 0; v < 8; ++v) {
    float m2 = lmax2[v], ss = lsum[v], bd = lmind2[v];
    int bi = lminid[v];
#pragma unroll
    for (int off = 1; off < 16; off <<= 1) {
      float om = __shfl_xor(m2, off, 32);
      float os = __shfl_xor(ss, off, 32);
      float nm = fmaxf(m2, om);
      ss = ss * __builtin_amdgcn_exp2f(m2 - nm) + os * __builtin_amdgcn_exp2f(om - nm);
      m2 = nm;
      float od = __shfl_xor(bd, off, 32);
      int   oi = __shfl_xor(bi, off, 32);
      bool bt = (od < bd) || (od == bd && oi < bi);
      bd = bt ? od : bd;
      bi = bt ? oi : bi;
    }
    rowmax2[v] = m2;
    inv_rs[v]  = __builtin_amdgcn_rcpf(ss);
    minid[v]   = bi;
  }

  v8f soft[4] = {};
  float* probs_out = out + O_PROBS;

  // ================= sweep 2: write probs, accumulate avg & soft = P@C ===============
#if HAVE_ASYNC
  __syncthreads();                       // all waves out of sweep-1 buffers
  if (wave == 0) issue_chunk<true>(cbh_k, cbt_k, C_h[0], C_t[0], 0, lane);
#endif
  for (int c = 0; c < 32; ++c) {
#if HAVE_ASYNC
    __syncthreads();
    if (wave == 0) {
      if (c < 31) {
        issue_chunk<true>(cbh_k, cbt_k, C_h[(c + 1) & 1], C_t[(c + 1) & 1], c + 1, lane);
        __builtin_amdgcn_s_wait_asynccnt(16);
      } else {
        __builtin_amdgcn_s_wait_asynccnt(0);
      }
    }
    __syncthreads();
    _Float16 (*Ch)[72] = C_h[c & 1];
    _Float16 (*Ct)[40] = C_t[c & 1];
#else
    __syncthreads();
    for (int t = tid; t < 512; t += 256) {
      int row = t >> 4, c4 = (t & 15) * 4;
      float4 v = *(const float4*)(cbbase + (size_t)(c * 32 + row) * ND + c4);
      _Float16 h0 = (_Float16)v.x, h1 = (_Float16)v.y, h2 = (_Float16)v.z, h3 = (_Float16)v.w;
      h4v h = {h0, h1, h2, h3};
      *(h4v*)&C_h[0][row][c4] = h;
      C_t[0][c4 + 0][row] = h0; C_t[0][c4 + 1][row] = h1;
      C_t[0][c4 + 2][row] = h2; C_t[0][c4 + 3][row] = h3;
    }
    __syncthreads();
    _Float16 (*Ch)[72] = C_h[0];
    _Float16 (*Ct)[40] = C_t[0];
#endif
#pragma unroll
    for (int nt = 0; nt < 2; ++nt) {
      v8f acc = {};
#pragma unroll
      for (int kt = 0; kt < 2; ++kt) {
        v16h b = load_h16(&Ch[nt * 16 + r][kt * 32 + 16 * half]);
        acc = __builtin_amdgcn_wmma_f32_16x16x32_f16(false, afrag[kt], false, b,
                                                     (short)0, acc, false, false);
      }
      const int s0 = c * 32 + nt * 16;
      const float cs = csbase[s0 + r];
      float colsum = 0.f;
#pragma unroll
      for (int v = 0; v < 8; ++v) {
        float d2 = xsq_v[v] + cs - 2.0f * acc[v];
        float p  = __builtin_amdgcn_exp2f(NEG2_INVLN2 * d2 - rowmax2[v]) * inv_rs[v];
        probs_out[(size_t)(m0 + v + 8 * half) * (NK * NS) + (size_t)k * NS + s0 + r] = p;
        colsum += p * mask_v[v];
        p_st[wave][v + 8 * half][nt * 16 + r] = (_Float16)p;
      }
      colsum += __shfl_xor(colsum, 16, 32);
      if (lane < 16) atomAddShared(&avg_lds[s0 + lane], colsum);
    }
    // soft += P(16x32) @ C(32x64)
    v16h pa;
    {
      union { v16h v; h8v h[2]; } u;
      u.h[0] = *(const h8v*)&p_st[wave][r][16 * half];
      u.h[1] = *(const h8v*)&p_st[wave][r][16 * half + 8];
      pa = u.v;
    }
#pragma unroll
    for (int dt = 0; dt < 4; ++dt) {
      v16h bc = load_h16(&Ct[dt * 16 + r][16 * half]);
      soft[dt] = __builtin_amdgcn_wmma_f32_16x16x32_f16(false, pa, false, bc,
                                                        (short)0, soft[dt], false, false);
    }
  }

  __syncthreads();
  for (int i = tid; i < 1024; i += 256)
    atomAddGlobal(&ws[W_AVG + k * NS + i], avg_lds[i]);

  // losses: sum (x - soft)^2
  float lsumq = 0.f;
#pragma unroll
  for (int dt = 0; dt < 4; ++dt)
#pragma unroll
    for (int v = 0; v < 8; ++v) {
      size_t m = (size_t)(m0 + v + 8 * half);
      int dd = dt * 16 + r;
      float xv = lat[m * LDIM + k * ND + dd];
      float df = xv - soft[dt][v];
      lsumq += df * df;
    }
#pragma unroll
  for (int off = 1; off < 32; off <<= 1) lsumq += __shfl_xor(lsumq, off, 32);
  if (lane == 0) atomAddShared(&lds_sq, lsumq);

  // hard_q / st_q / symbol_ids
#pragma unroll
  for (int rr = 0; rr < 16; ++rr) {
    int id = __shfl(minid[rr & 7], (rr >> 3) * 16, 32);
    size_t m = (size_t)(m0 + rr);
    const float* crow = cbk + (size_t)(k * NS + id) * ND;
    float h0 = crow[lane * 2], h1 = crow[lane * 2 + 1];
    size_t ob = m * LDIM + (size_t)k * ND + lane * 2;
    out[O_HARD + ob] = h0; out[O_HARD + ob + 1] = h1;
    out[O_ST + ob]   = h0; out[O_ST + ob + 1]   = h1;
    if (lane == 0) out[O_SYM + m * NK + k] = (float)id;
  }

  __syncthreads();
  if (tid == 0) atomAddGlobal(&ws[W_SQ], lds_sq);
}

__global__ __launch_bounds__(256) void vq_fin(const float* __restrict__ mask,
                                              const float* __restrict__ ws,
                                              float* __restrict__ out) {
  __shared__ float red[8], uarr[8], harr[8];
  __shared__ float sden;
  int tid = threadIdx.x, lane = tid & 31, w = tid >> 5;
  float s = 0.f;
  for (int i = tid; i < NM; i += 256) s += mask[i];
#pragma unroll
  for (int off = 1; off < 32; off <<= 1) s += __shfl_xor(s, off, 32);
  if (lane == 0) red[w] = s;
  __syncthreads();
  if (tid == 0) {
    float t = 0.f;
    for (int i = 0; i < 8; ++i) t += red[i];
    sden = fmaxf(t, 1.0f);
  }
  __syncthreads();
  float inv_den = 1.0f / sden;
  float u = 0.f, h = 0.f;
  for (int s2 = lane; s2 < NS; s2 += 32) {
    float a = ws[W_AVG + w * NS + s2] * inv_den;
    u += a * (logf(fmaxf(a, 1e-8f)) + 6.9314718055994531f); // + log(1024)
    h += a * logf(a + 1e-8f);
  }
#pragma unroll
  for (int off = 1; off < 32; off <<= 1) {
    u += __shfl_xor(u, off, 32);
    h += __shfl_xor(h, off, 32);
  }
  if (lane == 0) { uarr[w] = u; harr[w] = h; }
  __syncthreads();
  if (tid == 0) {
    float um = 0.f, pm = 0.f;
    for (int i = 0; i < 8; ++i) { um += uarr[i]; pm += __expf(-harr[i]); }
    float sq = ws[W_SQ];
    out[O_SCAL + 0] = 0.25f * sq * (1.0f / 4194304.0f); // commit
    out[O_SCAL + 1] = sq * (1.0f / 4194304.0f);         // codebook
    out[O_SCAL + 2] = 0.1f * (um * 0.125f);             // usage
    out[O_SCAL + 3] = pm * 0.125f;                      // perplexity
  }
}

extern "C" void kernel_launch(void* const* d_in, const int* in_sizes, int n_in,
                              void* d_out, int out_size, void* d_ws, size_t ws_size,
                              hipStream_t stream) {
  const float* lat  = (const float*)d_in[0];
  const float* mask = (const float*)d_in[1];
  const float* cbk  = (const float*)d_in[2];
  float* out = (float*)d_out;
  float* ws  = (float*)d_ws;
  _Float16* cbh = (_Float16*)(ws + W_H16);
  _Float16* cbt = (_Float16*)(ws + W_T16);
  vq_zero<<<33, 256, 0, stream>>>(ws);
  vq_csq<<<32, 256, 0, stream>>>(cbk, ws + W_CSQ, cbh, cbt);
  vq_main<<<512, 256, 0, stream>>>(lat, mask, cbk, ws + W_CSQ, cbh, cbt, out, ws);
  vq_fin<<<1, 256, 0, stream>>>(mask, ws, out);
}
